// ComplexMixture_6133213299320
// MI455X (gfx1250) — matrix operand order repaired and checked by reading, not compiled, not run on hardware
//
#include <hip/hip_runtime.h>

typedef __bf16 bf16_t;
typedef bf16_t v16bf __attribute__((ext_vector_type(16)));
typedef bf16_t v8bf  __attribute__((ext_vector_type(8)));
typedef float  v8f   __attribute__((ext_vector_type(8)));

#define WMMA_BF16(A, B, C) \
  __builtin_amdgcn_wmma_f32_16x16x32_bf16(false, (A), false, (B), (short)0, (C), false, false)

constexpr int Bx = 4;
constexpr int Sx = 8192;
constexpr int Dx = 512;
constexpr int BM = 64;        // d-block
constexpr int BN = 64;        // e-block
constexpr int BK = 32;        // K step (one WMMA K)
constexpr int NK = Sx / BK;   // 256 iterations
constexpr int NT = 256;       // threads per block (8 waves)
constexpr int LDK = 40;       // padded K stride in halves (80B rows, conflict-free 16B lanes)
constexpr int TILE = BM * LDK;          // 2560 halves = 5120 B per tile
constexpr int AREG = 4 * TILE;          // B region starts at 20480 B
// LDS: A region tiles {R_hi,R_lo,I_hi,I_lo} then B region same order => 40 KB.
// Every DS access = one of 4 thread-base VGPRs + immediate (< 40 KB, all legal).

__global__ __launch_bounds__(NT)
void cgram_bf16x3_kernel(const float* __restrict__ R,
                         const float* __restrict__ I,
                         const float* __restrict__ W,
                         float* __restrict__ outR,
                         float* __restrict__ outI)
{
    __shared__ bf16_t lds[8 * TILE];

    const int te   = blockIdx.x;          // e-block (0..7)
    const int td   = blockIdx.y;          // d-block (0..7)
    const int b    = blockIdx.z;          // batch  (0..3)
    const int t    = threadIdx.x;         // 0..255
    const int lane = t & 31;
    const int wid  = t >> 5;              // 0..7
    const int mw   = wid & 1;             // wave row (0..1) -> 32 rows each
    const int nw   = wid >> 1;            // wave col (0..3) -> 16 cols each

    // ---- cooperative tile-load mapping: column c, 8 consecutive s values ----
    const int c  = t & 63;                // column within 64-wide block
    const int kq = t >> 6;                // 0..3
    const int k4 = kq * 8;                // this thread's k offset within BK

    // running global pointers (bumped by BK*Dx per K-step)
    const size_t gbase = (size_t)b * Sx * Dx + (size_t)k4 * Dx;
    const float* pRA = R + gbase + td * BM + c;
    const float* pIA = I + gbase + td * BM + c;
    const float* pRB = R + gbase + te * BN + c;
    const float* pIB = I + gbase + te * BN + c;
    const float* pW  = W + (size_t)b * Sx + k4;

    float ar[8], ai[8], br[8], bi[8];

    // weight folded into B-side at load staging
    auto load_tile = [&]() {
#pragma unroll
        for (int j = 0; j < 8; ++j) {
            const float w = pW[j];
            ar[j] = pRA[j * Dx];
            ai[j] = pIA[j * Dx];
            br[j] = pRB[j * Dx] * w;
            bi[j] = pIB[j * Dx] * w;
        }
        pRA += BK * Dx; pIA += BK * Dx; pRB += BK * Dx; pIB += BK * Dx; pW += BK;
    };

    // cheap truncation split: hi = top 16 bits of f32, lo = rne-bf16(x - hi)
    auto split8 = [&](const float* v, v8bf& h, v8bf& l) {
#pragma unroll
        for (int j = 0; j < 8; ++j) {
            union { float f; unsigned u; } x;  x.f = v[j];
            union { float f; unsigned u; } hf; hf.u = x.u & 0xFFFF0000u;
            union { unsigned short s; bf16_t b; } hb;
            hb.s = (unsigned short)(x.u >> 16);
            h[j] = hb.b;
            l[j] = (bf16_t)(v[j] - hf.f);
        }
    };

    // thread base pointers (4 address VGPRs, everything else is immediates)
    bf16_t*       stA = &lds[c * LDK + k4];                 // A-region store base
    bf16_t*       stB = stA + AREG;                         // B-region store base
    const int k0  = (lane < 16) ? 0 : 8;                    // A-frag K offset
    const int kb0 = (lane < 16) ? 0 : 8;                    // B-frag K offset
    const bf16_t* fA = &lds[(mw * 32 + (lane & 15)) * LDK + k0];
    const bf16_t* fB = &lds[AREG + (nw * 16 + (lane & 15)) * LDK + kb0];

    // A fragment (16x32): lanes 0-15 -> M=lane, K={0..7,16..23}; 16-31 -> K={8..15,24..31}
    auto load_afrag = [&](int tile, int mt) -> v16bf {
        const bf16_t* p = fA + tile * TILE + mt * 16 * LDK;
        v8bf a0 = *(const v8bf*)(p);
        v8bf a1 = *(const v8bf*)(p + 16);
        return __builtin_shufflevector(a0, a1, 0,1,2,3,4,5,6,7,8,9,10,11,12,13,14,15);
    };
    // B fragment (32x16): lanes 0-15 -> N=lane,K={0..7,16..23}; 16-31 -> K={8..15,24..31} ([col][k] tile)
    auto load_bfrag = [&](int tile) -> v16bf {
        const bf16_t* p = fB + tile * TILE;
        v8bf b0 = *(const v8bf*)(p);
        v8bf b1 = *(const v8bf*)(p + 16);
        return __builtin_shufflevector(b0, b1, 0,1,2,3,4,5,6,7,8,9,10,11,12,13,14,15);
    };

    v8f acc_rr[2] = {};
    v8f acc_ii[2] = {};
    v8f acc_ir[2] = {};
    v8f acc_ri[2] = {};

    load_tile();

    for (int it = 0; it < NK; ++it) {
        __syncthreads();   // previous compute done: safe to overwrite LDS

        // split hi/lo, fill LDS (b128 stores)
        {
            v8bf h, l;
            split8(ar, h, l); *(v8bf*)(stA + 0*TILE) = h; *(v8bf*)(stA + 1*TILE) = l;
            split8(ai, h, l); *(v8bf*)(stA + 2*TILE) = h; *(v8bf*)(stA + 3*TILE) = l;
            split8(br, h, l); *(v8bf*)(stB + 0*TILE) = h; *(v8bf*)(stB + 1*TILE) = l;
            split8(bi, h, l); *(v8bf*)(stB + 2*TILE) = h; *(v8bf*)(stB + 3*TILE) = l;
        }

        // prefetch next K-slab into registers (overlaps barrier + compute)
        if (it + 1 < NK) load_tile();

        __syncthreads();   // tile visible to all waves

        // B-side fragments for this wave's single n-tile (R/I x hi/lo)
        v16bf rbh = load_bfrag(0);
        v16bf rbl = load_bfrag(1);
        v16bf ibh = load_bfrag(2);
        v16bf ibl = load_bfrag(3);

        // per m-tile: load A fragments, run 12 WMMAs (short A lifetimes)
#pragma unroll
        for (int mt = 0; mt < 2; ++mt) {
            v16bf rah = load_afrag(0, mt);
            v16bf ral = load_afrag(1, mt);
            v16bf iah = load_afrag(2, mt);
            v16bf ial = load_afrag(3, mt);

            // bf16x3: hi*hi + hi*lo + lo*hi
            acc_rr[mt] = WMMA_BF16(rah, rbh, acc_rr[mt]);
            acc_rr[mt] = WMMA_BF16(rah, rbl, acc_rr[mt]);
            acc_rr[mt] = WMMA_BF16(ral, rbh, acc_rr[mt]);

            acc_ii[mt] = WMMA_BF16(iah, ibh, acc_ii[mt]);
            acc_ii[mt] = WMMA_BF16(iah, ibl, acc_ii[mt]);
            acc_ii[mt] = WMMA_BF16(ial, ibh, acc_ii[mt]);

            acc_ir[mt] = WMMA_BF16(iah, rbh, acc_ir[mt]);
            acc_ir[mt] = WMMA_BF16(iah, rbl, acc_ir[mt]);
            acc_ir[mt] = WMMA_BF16(ial, rbh, acc_ir[mt]);

            acc_ri[mt] = WMMA_BF16(rah, ibh, acc_ri[mt]);
            acc_ri[mt] = WMMA_BF16(rah, ibl, acc_ri[mt]);
            acc_ri[mt] = WMMA_BF16(ral, ibh, acc_ri[mt]);
        }
    }

    // ---- epilogue: C/D layout = VGPR j -> M=j (lanes 0-15), M=j+8 (lanes 16-31)
    const size_t DD = (size_t)Dx * Dx;
    float* oR = outR + (size_t)b * DD;
    float* oI = outI + (size_t)b * DD;
    const int rhalf = (lane >> 4) * 8;
    const int coln  = te * BN + nw * 16 + (lane & 15);

#pragma unroll
    for (int mt = 0; mt < 2; ++mt) {
#pragma unroll
        for (int j = 0; j < 8; ++j) {
            const int row = td * BM + mw * 32 + mt * 16 + rhalf + j;
            oR[(size_t)row * Dx + coln] = acc_rr[mt][j] + acc_ii[mt][j];
            oI[(size_t)row * Dx + coln] = acc_ir[mt][j] - acc_ri[mt][j];
        }
    }
}

extern "C" void kernel_launch(void* const* d_in, const int* in_sizes, int n_in,
                              void* d_out, int out_size, void* d_ws, size_t ws_size,
                              hipStream_t stream) {
    const float* R = (const float*)d_in[0];   // input_real  [B,S,D]
    const float* I = (const float*)d_in[1];   // input_imag  [B,S,D]
    const float* W = (const float*)d_in[2];   // weight      [B,S]
    float* outR = (float*)d_out;                               // [B,D,D]
    float* outI = outR + (size_t)Bx * Dx * Dx;                 // [B,D,D]

    dim3 grid(Dx / BN, Dx / BM, Bx);   // (e-blocks, d-blocks, batch) = (8,8,4) = 256 WGs
    cgram_bf16x3_kernel<<<grid, NT, 0, stream>>>(R, I, W, outR, outI);
}
